// AttentionBlock_9517647527960
// MI455X (gfx1250) — compile-verified, matrix operand order checked by reading, not run on hardware
//
#include <hip/hip_runtime.h>
#include <hip/hip_bf16.h>
#include <math.h>

typedef __attribute__((ext_vector_type(16))) _Float16 v16h;
typedef __attribute__((ext_vector_type(8)))  _Float16 h8;
typedef __attribute__((ext_vector_type(4)))  _Float16 h4;
typedef __attribute__((ext_vector_type(8)))  float    v8f;
typedef __attribute__((ext_vector_type(4)))  unsigned int u32x4;
typedef __attribute__((ext_vector_type(8)))  int      i32x8;
typedef __attribute__((ext_vector_type(4)))  int      i32x4;

#define LQ    4096      // H*W
#define CCH   256       // channels
#define HD    64        // head dim
#define NCOLS 8192      // N * L

// ---------------------------------------------------------------------------
// WMMA helper (CDNA5 v_wmma_f32_16x16x32_f16, wave32)
// ---------------------------------------------------------------------------
__device__ __forceinline__ v8f wmma_f32_16x16x32(v16h a, v16h b, v8f c) {
  return __builtin_amdgcn_wmma_f32_16x16x32_f16(
      false, a, false, b, (short)0, c, false, false);
}

// Fragment load: element (x, k) stored at s[x*ld + k] where x is the lane
// index (M for A-frags, N for B-frags). K-pattern per ISA 7.12.2:
// halves 0..7 = K base+0..7, halves 8..15 = K base+16..23, base = (lane>>4)*8.
// Two 16-byte LDS reads per fragment (ld*2 bytes must be a multiple of 16).
__device__ __forceinline__ v16h ldfrag(const _Float16* s, int ld, int lane) {
  const _Float16* p = s + (lane & 15) * ld + (((lane >> 4) & 1) << 3);
  h8 lo = *(const h8*)(p);
  h8 hi = *(const h8*)(p + 16);
  return __builtin_shufflevector(lo, hi, 0,1,2,3,4,5,6,7,8,9,10,11,12,13,14,15);
}

__device__ __forceinline__ void transpose4x4(h4 r0, h4 r1, h4 r2, h4 r3,
                                             h4& c0, h4& c1, h4& c2, h4& c3) {
  c0 = h4{r0[0], r1[0], r2[0], r3[0]};
  c1 = h4{r0[1], r1[1], r2[1], r3[1]};
  c2 = h4{r0[2], r1[2], r2[2], r3[2]};
  c3 = h4{r0[3], r1[3], r2[3], r3[3]};
}

// ---------------------------------------------------------------------------
// Tensor Data Mover: 2D tile load, global -> LDS.
// clang-23 / amdgpu-toolchain 6-arg builtin:
//   (uint32x4 g0, int32x8 g1, int32x4 g2, int32x4 g3, int32x8, i32 cpol)
// D# packing per CDNA5 ISA. data_size = 2 bytes.
// pad codes: interval 0=2DW,1=4,2=8,3=16,4=32...; amount 0=1DW,...,3=4DW.
// ---------------------------------------------------------------------------
__device__ __forceinline__ unsigned lds_offset_of(const void* p) {
  return (unsigned)(unsigned long long)(__attribute__((address_space(3))) const void*)p;
}

__device__ __forceinline__ void tdm_load_2d_f16(
    const void* gaddr, unsigned lds_byte_off,
    unsigned tile_d0, unsigned tile_d1, unsigned long long stride_elems,
    unsigned pad_interval_code, unsigned pad_amount_code) {
  unsigned long long ga = (unsigned long long)gaddr;
  u32x4 g0;
  g0[0] = 1u;                                   // count=1, user descriptor
  g0[1] = lds_byte_off;                         // lds_addr
  g0[2] = (unsigned)ga;                         // global_addr[31:0]
  g0[3] = (unsigned)(ga >> 32) | (2u << 30);    // global_addr[56:32] | type=2
  const unsigned td0 = 1u << 20, td1 = 1u << 20;  // generous dims: no OOB clip
  i32x8 g1;
  g1[0] = (int)((1u << 16)                      // data_size = 2 bytes
              | (1u << 20)                      // pad_enable
              | (pad_interval_code << 22)
              | (pad_amount_code << 25));
  g1[1] = (int)((td0 & 0xFFFFu) << 16);         // [47:32]=0, [63:48]=td0.lo
  g1[2] = (int)(((td0 >> 16) & 0xFFFFu) | ((td1 & 0xFFFFu) << 16));
  g1[3] = (int)(((td1 >> 16) & 0xFFFFu) | (tile_d0 << 16));   // tile_dim0
  g1[4] = (int)(tile_d1 & 0xFFFFu);                            // tile_dim1
  g1[5] = (int)(unsigned)(stride_elems & 0xFFFFFFFFull);       // dim0 stride lo
  g1[6] = (int)(unsigned)((stride_elems >> 32) & 0xFFFFull);   // stride hi
  g1[7] = 0;
  i32x4 z4 = {0, 0, 0, 0};
  i32x8 z8 = {0, 0, 0, 0, 0, 0, 0, 0};
  __builtin_amdgcn_tensor_load_to_lds(g0, g1, z4, z4, z8, 0);
}

// ---------------------------------------------------------------------------
// Weight conversion f32 -> f16
// ---------------------------------------------------------------------------
__global__ void __launch_bounds__(256) cvt_weights_kernel(
    const float* __restrict__ wq, const float* __restrict__ wk,
    const float* __restrict__ wv, const float* __restrict__ wp,
    _Float16* __restrict__ dq, _Float16* __restrict__ dk,
    _Float16* __restrict__ dv, _Float16* __restrict__ dp) {
  int i = blockIdx.x * 256 + threadIdx.x;
  if (i < CCH * CCH) {
    dq[i] = (_Float16)wq[i];
    dk[i] = (_Float16)wk[i];
    dv[i] = (_Float16)wv[i];
    dp[i] = (_Float16)wp[i];
  }
}

// ---------------------------------------------------------------------------
// GroupNorm: one block per (n, group); writes f16 activation in [C][N*L].
// ---------------------------------------------------------------------------
__global__ void __launch_bounds__(256) groupnorm_kernel(
    const float* __restrict__ x, const float* __restrict__ gw,
    const float* __restrict__ gb, _Float16* __restrict__ xh) {
  __shared__ float s_sum[256], s_sq[256];
  int g = blockIdx.x;
  int n = g >> 3, grp = g & 7;
  int cbase = grp * 32;
  const float* xg = x + ((size_t)n * CCH + cbase) * LQ;
  int t = threadIdx.x;
  float sum = 0.f, sq = 0.f;
  for (int i = t; i < 32 * LQ; i += 256) {
    float v = xg[i];
    sum += v; sq += v * v;
  }
  s_sum[t] = sum; s_sq[t] = sq;
  __syncthreads();
  for (int s = 128; s > 0; s >>= 1) {
    if (t < s) { s_sum[t] += s_sum[t + s]; s_sq[t] += s_sq[t + s]; }
    __syncthreads();
  }
  const float inv_n = 1.f / (32.f * (float)LQ);
  float mu   = s_sum[0] * inv_n;
  float var  = s_sq[0] * inv_n - mu * mu;
  float rstd = rsqrtf(var + 1e-5f);
  for (int i = t; i < 32 * LQ; i += 256) {
    int c = cbase + (i >> 12);
    int l = i & 4095;
    float v = (xg[i] - mu) * rstd * gw[c] + gb[c];
    xh[(size_t)c * NCOLS + (size_t)n * LQ + l] = (_Float16)v;
  }
}

// ---------------------------------------------------------------------------
// GEMM  Y[256 x 8192] = A[256 x 256] * B[256 x 8192]  (+bias epilogue)
// block tile 128x128, 8 waves as 2x4, wave tile 64x32 (4x2 WMMA frags).
// A tile: TDM async load (natural layout). B tile: staged TRANSPOSED so
// B-fragments are two ds_load_b128 each.
// ---------------------------------------------------------------------------
template <int MODE>
__global__ void __launch_bounds__(256) gemm_kernel(
    const _Float16* __restrict__ A, const _Float16* __restrict__ B,
    const float* __restrict__ bias, const float* __restrict__ xres,
    float scale, _Float16* __restrict__ Yh, float* __restrict__ Yf) {
  __shared__ __align__(16) _Float16 As[128 * 40];   // [d][k], 32 K + 8 pad
  __shared__ __align__(16) _Float16 Bt[128 * 40];   // [n][k] transposed
  int t = threadIdx.x;
  int lane = t & 31, wave = t >> 5;
  int dBase   = blockIdx.x * 128;
  int colBase = blockIdx.y * 128;
  int wm = wave >> 2, wn = wave & 3;

  v8f acc[4][2] = {};

  for (int k0 = 0; k0 < CCH; k0 += 32) {
    // async DMA of A tile (128 rows x 32 halfs, row pitch 256) into As.
    // pad: 32 halfs = 16 DWORDs -> interval code 3; 8-half pad = 4 DW -> code 3
    if (wave == 0) {
      tdm_load_2d_f16(&A[(size_t)dBase * CCH + k0], lds_offset_of(As),
                      /*tile_d0=*/32, /*tile_d1=*/128, /*stride=*/CCH, 3, 3);
    }
    // stage B tile transposed: each thread owns a 4x4 block
    {
      int kb = (t >> 5) * 4;          // 0..28
      int nb = (t & 31) * 4;          // 0..124
      const _Float16* gp = &B[(size_t)(k0 + kb) * NCOLS + colBase + nb];
      h4 r0 = *(const h4*)(gp);
      h4 r1 = *(const h4*)(gp + NCOLS);
      h4 r2 = *(const h4*)(gp + 2 * NCOLS);
      h4 r3 = *(const h4*)(gp + 3 * NCOLS);
      h4 c0, c1, c2, c3;
      transpose4x4(r0, r1, r2, r3, c0, c1, c2, c3);
      *(h4*)&Bt[(nb + 0) * 40 + kb] = c0;
      *(h4*)&Bt[(nb + 1) * 40 + kb] = c1;
      *(h4*)&Bt[(nb + 2) * 40 + kb] = c2;
      *(h4*)&Bt[(nb + 3) * 40 + kb] = c3;
    }
    if (wave == 0) __builtin_amdgcn_s_wait_tensorcnt(0);
    __syncthreads();

    v16h a[4], b[2];
#pragma unroll
    for (int i = 0; i < 4; ++i)
      a[i] = ldfrag(&As[(wm * 64 + i * 16) * 40], 40, lane);
#pragma unroll
    for (int j = 0; j < 2; ++j)
      b[j] = ldfrag(&Bt[(wn * 32 + j * 16) * 40], 40, lane);
#pragma unroll
    for (int i = 0; i < 4; ++i)
#pragma unroll
      for (int j = 0; j < 2; ++j)
        acc[i][j] = wmma_f32_16x16x32(a[i], b[j], acc[i][j]);
    __syncthreads();
  }

  // epilogue (D layout: lane&15 = col, row = ((lane>>4)<<3)+r)
#pragma unroll
  for (int i = 0; i < 4; ++i) {
#pragma unroll
    for (int j = 0; j < 2; ++j) {
      int col = colBase + wn * 32 + j * 16 + (lane & 15);
#pragma unroll
      for (int r = 0; r < 8; ++r) {
        int d = dBase + wm * 64 + i * 16 + ((lane >> 4) << 3) + r;
        float v = acc[i][j][r] + bias[d];
        if (MODE == 0) {
          Yh[(size_t)d * NCOLS + col] = (_Float16)(v * scale);
        } else {
          int n = col >> 12, l = col & 4095;
          size_t idx = ((size_t)n * CCH + d) * LQ + l;
          Yf[idx] = xres[idx] + v;
        }
      }
    }
  }
}

// ---------------------------------------------------------------------------
// Flash attention. grid = (64 q-tiles, N*heads). Q/K/V layout [d][N*L] f16,
// q pre-scaled by hd^-0.5. Tiles staged so EVERY fragment is 2x ds_load_b128:
//   Qs[q][c] (transposed once), Kt[kl][c] (transposed per tile),
//   Vs[c][kl] (natural -> loaded by the Tensor Data Mover), Ps[q][kl].
// ---------------------------------------------------------------------------
__global__ void __launch_bounds__(256) flash_attn_kernel(
    const _Float16* __restrict__ Q, const _Float16* __restrict__ K,
    const _Float16* __restrict__ V, _Float16* __restrict__ O) {
  __shared__ __align__(16) _Float16 Qs[64 * 72];   // [q][c]
  __shared__ __align__(16) _Float16 Kt[64 * 72];   // [kl][c]
  __shared__ __align__(16) _Float16 Vs[64 * 72];   // [c][kl]  (TDM target)
  __shared__ __align__(16) _Float16 Ps[64 * 72];   // [q][kl]
  __shared__ float Ss[64 * 68];                    // [q][kl] f32 scores
  __shared__ float mS[64], lS[64], aS[64];

  int t = threadIdx.x;
  int lane = t & 31, wave = t >> 5;
  int bq = blockIdx.x;
  int nh = blockIdx.y;
  int n = nh >> 2, h = nh & 3;
  size_t headOff = (size_t)(h * HD) * NCOLS + (size_t)n * LQ;
  const _Float16* Qp = Q + headOff + bq * 64;
  const _Float16* Kp = K + headOff;
  const _Float16* Vp = V + headOff;

  {   // Q tile (global [c][q]) -> Qs[q][c] via 4x4 register transpose
    int cb = (t >> 4) * 4;
    int qb = (t & 15) * 4;
    const _Float16* gp = &Qp[(size_t)cb * NCOLS + qb];
    h4 r0 = *(const h4*)(gp);
    h4 r1 = *(const h4*)(gp + NCOLS);
    h4 r2 = *(const h4*)(gp + 2 * NCOLS);
    h4 r3 = *(const h4*)(gp + 3 * NCOLS);
    h4 c0, c1, c2, c3;
    transpose4x4(r0, r1, r2, r3, c0, c1, c2, c3);
    *(h4*)&Qs[(qb + 0) * 72 + cb] = c0;
    *(h4*)&Qs[(qb + 1) * 72 + cb] = c1;
    *(h4*)&Qs[(qb + 2) * 72 + cb] = c2;
    *(h4*)&Qs[(qb + 3) * 72 + cb] = c3;
  }
  if (t < 64) { mS[t] = -3.0e38f; lS[t] = 0.f; }

  v8f accO[2] = {};
  int qs = (wave >> 1) * 16;       // wave's q strip
  int xs = (wave & 1) * 32;        // wave's kl (S) / c (PV) column offset
  __syncthreads();

  for (int kt = 0; kt < LQ; kt += 64) {
    // V tile (64c x 64kl, row pitch NCOLS) -> Vs via TDM.
    // pad: 64 halfs = 32 DWORDs -> interval code 4; 8-half pad -> code 3
    if (wave == 0) {
      tdm_load_2d_f16(&Vp[kt], lds_offset_of(Vs),
                      /*tile_d0=*/64, /*tile_d1=*/64, /*stride=*/NCOLS, 4, 3);
    }
    {   // K tile (global [c][kl]) -> Kt[kl][c] via 4x4 transpose
      int cb = (t >> 4) * 4;
      int kb = (t & 15) * 4;
      const _Float16* gp = &Kp[(size_t)cb * NCOLS + kt + kb];
      h4 r0 = *(const h4*)(gp);
      h4 r1 = *(const h4*)(gp + NCOLS);
      h4 r2 = *(const h4*)(gp + 2 * NCOLS);
      h4 r3 = *(const h4*)(gp + 3 * NCOLS);
      h4 c0, c1, c2, c3;
      transpose4x4(r0, r1, r2, r3, c0, c1, c2, c3);
      *(h4*)&Kt[(kb + 0) * 72 + cb] = c0;
      *(h4*)&Kt[(kb + 1) * 72 + cb] = c1;
      *(h4*)&Kt[(kb + 2) * 72 + cb] = c2;
      *(h4*)&Kt[(kb + 3) * 72 + cb] = c3;
    }
    if (wave == 0) __builtin_amdgcn_s_wait_tensorcnt(0);
    __syncthreads();

    // S = Q^T K : A frag (m=q,k=c) from Qs[q][c]; B frag (n=kl,k=c) from Kt
    v8f accS[2] = {};
#pragma unroll
    for (int kk = 0; kk < HD; kk += 32) {
      v16h a = ldfrag(&Qs[qs * 72 + kk], 72, lane);
#pragma unroll
      for (int j = 0; j < 2; ++j) {
        v16h b = ldfrag(&Kt[(xs + j * 16) * 72 + kk], 72, lane);
        accS[j] = wmma_f32_16x16x32(a, b, accS[j]);
      }
    }
#pragma unroll
    for (int j = 0; j < 2; ++j)
#pragma unroll
      for (int r = 0; r < 8; ++r)
        Ss[(qs + ((lane >> 4) << 3) + r) * 68 + xs + j * 16 + (lane & 15)] =
            accS[j][r];
    __syncthreads();

    {   // online softmax: 4 threads per q-row, 16 cols each
      int row = t >> 2, t4 = t & 3;
      float vals[16];
      float vmax = -3.0e38f;
#pragma unroll
      for (int i = 0; i < 16; ++i) {
        vals[i] = Ss[row * 68 + t4 * 16 + i];
        vmax = fmaxf(vmax, vals[i]);
      }
      vmax = fmaxf(vmax, __shfl_xor(vmax, 1, 32));
      vmax = fmaxf(vmax, __shfl_xor(vmax, 2, 32));
      float mold = mS[row];
      float mnew = fmaxf(mold, vmax);
      float sum = 0.f;
#pragma unroll
      for (int i = 0; i < 16; ++i) {
        float p = __expf(vals[i] - mnew);
        sum += p;
        Ps[row * 72 + t4 * 16 + i] = (_Float16)p;
      }
      sum += __shfl_xor(sum, 1, 32);
      sum += __shfl_xor(sum, 2, 32);
      if (t4 == 0) {
        float alpha = __expf(mold - mnew);
        aS[row] = alpha;
        mS[row] = mnew;
        lS[row] = lS[row] * alpha + sum;
      }
    }
    __syncthreads();

    // rescale running O, then O += P * V^T
    // A frag (m=q,k=kl) from Ps[q][kl]; B frag (n=c,k=kl) from Vs[c][kl]
#pragma unroll
    for (int j = 0; j < 2; ++j)
#pragma unroll
      for (int r = 0; r < 8; ++r)
        accO[j][r] *= aS[qs + ((lane >> 4) << 3) + r];
#pragma unroll
    for (int kk = 0; kk < 64; kk += 32) {
      v16h a = ldfrag(&Ps[qs * 72 + kk], 72, lane);
#pragma unroll
      for (int j = 0; j < 2; ++j) {
        v16h b = ldfrag(&Vs[(xs + j * 16) * 72 + kk], 72, lane);
        accO[j] = wmma_f32_16x16x32(a, b, accO[j]);
      }
    }
    __syncthreads();
  }

  // finalize: divide by running sum, store [h*64+c][n*L + bq*64 + q] f16
#pragma unroll
  for (int j = 0; j < 2; ++j) {
    int c = xs + j * 16 + (lane & 15);
#pragma unroll
    for (int r = 0; r < 8; ++r) {
      int q = qs + ((lane >> 4) << 3) + r;
      float inv = 1.0f / lS[q];
      O[(size_t)(h * HD + c) * NCOLS + (size_t)n * LQ + bq * 64 + q] =
          (_Float16)(accO[j][r] * inv);
    }
  }
}

// ---------------------------------------------------------------------------
// Mask passthrough (second tuple output), int -> float
// ---------------------------------------------------------------------------
__global__ void __launch_bounds__(256) mask_copy_kernel(
    const int* __restrict__ m, float* __restrict__ o, int nelem) {
  int i = blockIdx.x * 256 + threadIdx.x;
  if (i < nelem) o[i] = (float)m[i];
}

// ---------------------------------------------------------------------------
extern "C" void kernel_launch(void* const* d_in, const int* in_sizes, int n_in,
                              void* d_out, int out_size, void* d_ws, size_t ws_size,
                              hipStream_t stream) {
  const float* x      = (const float*)d_in[0];
  const int*   mask   = (const int*)  d_in[1];
  const float* norm_w = (const float*)d_in[2];
  const float* norm_b = (const float*)d_in[3];
  const float* q_w = (const float*)d_in[4];
  const float* q_b = (const float*)d_in[5];
  const float* k_w = (const float*)d_in[6];
  const float* k_b = (const float*)d_in[7];
  const float* v_w = (const float*)d_in[8];
  const float* v_b = (const float*)d_in[9];
  const float* p_w = (const float*)d_in[10];
  const float* p_b = (const float*)d_in[11];

  _Float16* ws   = (_Float16*)d_ws;
  _Float16* wq16 = ws;
  _Float16* wk16 = wq16 + 65536;
  _Float16* wv16 = wk16 + 65536;
  _Float16* wp16 = wv16 + 65536;
  _Float16* xh   = wp16 + 65536;
  _Float16* qh   = xh + (size_t)CCH * NCOLS;
  _Float16* kh   = qh + (size_t)CCH * NCOLS;
  _Float16* vh   = kh + (size_t)CCH * NCOLS;
  _Float16* ah   = vh + (size_t)CCH * NCOLS;

  float* out      = (float*)d_out;
  float* out_mask = out + (size_t)2 * CCH * LQ;

  cvt_weights_kernel<<<256, 256, 0, stream>>>(q_w, k_w, v_w, p_w,
                                              wq16, wk16, wv16, wp16);
  groupnorm_kernel<<<16, 256, 0, stream>>>(x, norm_w, norm_b, xh);

  dim3 ggrid(2, 64);
  gemm_kernel<0><<<ggrid, 256, 0, stream>>>(wq16, xh, q_b, nullptr, 0.125f, qh, nullptr);
  gemm_kernel<0><<<ggrid, 256, 0, stream>>>(wk16, xh, k_b, nullptr, 1.0f,   kh, nullptr);
  gemm_kernel<0><<<ggrid, 256, 0, stream>>>(wv16, xh, v_b, nullptr, 1.0f,   vh, nullptr);

  flash_attn_kernel<<<dim3(64, 8), 256, 0, stream>>>(qh, kh, vh, ah);

  gemm_kernel<1><<<ggrid, 256, 0, stream>>>(wp16, ah, p_b, x, 1.0f, nullptr, out);
  mask_copy_kernel<<<32, 256, 0, stream>>>(mask, out_mask, 2 * LQ);
}
